// TransformerEncoderBlock_59107339927992
// MI455X (gfx1250) — matrix operand order repaired and checked by reading, not compiled, not run on hardware
//
#include <hip/hip_runtime.h>
#include <cmath>

typedef __bf16 bf16_t;
typedef __attribute__((ext_vector_type(16))) __bf16 v16bf;
typedef __attribute__((ext_vector_type(8)))  float  v8f;
typedef __attribute__((ext_vector_type(4)))  unsigned int u32x4;
typedef __attribute__((ext_vector_type(8)))  int i32x8;
typedef __attribute__((ext_vector_type(4)))  int i32x4;

#define D_MODEL 2048
#define N_HEADS 16
#define FF_DIM  8192
#define HEAD_DIM 128
#define BB 8
#define SS 1024
#define MROWS (BB*SS)          // 8192
#define QKV_N (3*D_MODEL)      // 6144

#if defined(__HIP_DEVICE_COMPILE__) && __has_builtin(__builtin_amdgcn_tensor_load_to_lds) && __has_builtin(__builtin_amdgcn_s_wait_tensorcnt)
#define HAVE_TDM 1
#else
#define HAVE_TDM 0
#endif

// ---------------- WMMA helpers ----------------
__device__ __forceinline__ v8f wmma_bf16(v16bf a, v16bf b, v8f c) {
  return __builtin_amdgcn_wmma_f32_16x16x32_bf16(false, a, false, b, (short)0, c, false, false);
}

// A-matrix 16x32 bf16 fragment: lane holds row (lane&15); K = kh..kh+7 and kh+16..kh+23, kh = (lane>>4)*8
__device__ __forceinline__ v16bf frag_a(const bf16_t* p, int lane) {
  const int kh = (lane >> 4) << 3;
  union { uint4 u[2]; v16bf v; } r;
  r.u[0] = *(const uint4*)(p + kh);
  r.u[1] = *(const uint4*)(p + kh + 16);
  return r.v;
}
// B-matrix 32x16 bf16 fragment: lane holds col (lane&15); K = kh..kh+7 and kh+16..kh+23, kh = (lane>>4)*8
__device__ __forceinline__ v16bf frag_b(const bf16_t* p, int lane) {
  const int kh = (lane >> 4) << 3;
  union { uint4 u[2]; v16bf v; } r;
  r.u[0] = *(const uint4*)(p + kh);
  r.u[1] = *(const uint4*)(p + kh + 16);
  return r.v;
}

__device__ __forceinline__ v8f v8f_zero() {
  v8f z = {0.f,0.f,0.f,0.f,0.f,0.f,0.f,0.f};
  return z;
}

__device__ __forceinline__ float gelu_exact(float x) {
  return 0.5f * x * (1.0f + erff(x * 0.70710678118654752f));
}

#if HAVE_TDM
// Get workgroup-relative LDS byte offset of a __shared__ object.
__device__ __forceinline__ unsigned lds_off(const void* p) {
  return (unsigned)(unsigned long long)(p);
}
// TDM 2-D tile load: bf16 (data_size=2B) tile [tile_d1 rows x tile_d0 elems] from a
// row-major tensor with row stride stride0 (elements) into LDS at lds_addr, with
// per-row LDS padding (pad codes per D# spec: interval 2^(p+1) DWORDs, amount a+1 DWORDs).
__device__ __forceinline__ void tdm_load_2d(const void* gptr, unsigned lds_addr,
                                            unsigned tile_d0, unsigned tile_d1,
                                            unsigned long long stride0,
                                            unsigned tensor_d0, unsigned tensor_d1,
                                            unsigned pad_int, unsigned pad_amt) {
  unsigned long long ga = (unsigned long long)gptr;
  u32x4 g0;
  g0[0] = 1u;                                               // count=1 (valid user D#)
  g0[1] = lds_addr;                                         // lds_addr [63:32]
  g0[2] = (unsigned)(ga & 0xFFFFFFFFull);                   // global_addr[31:0]
  g0[3] = (unsigned)((ga >> 32) & 0x01FFFFFFull) | (2u<<30);// global_addr[56:32], type=2
  i32x8 g1;
  g1[0] = (int)((1u << 16)            // data_size = 2 bytes
              | (1u << 20)            // pad_enable
              | (pad_int << 22)
              | (pad_amt << 25));
  g1[1] = (int)((tensor_d0 & 0xFFFFu) << 16);               // tensor_dim0[15:0] @ [63:48]
  g1[2] = (int)((tensor_d0 >> 16) | ((tensor_d1 & 0xFFFFu) << 16));
  g1[3] = (int)((tensor_d1 >> 16) | (tile_d0 << 16));       // tile_dim0 @ [127:112]
  g1[4] = (int)(tile_d1);                                   // tile_dim1, tile_dim2=0
  g1[5] = (int)(stride0 & 0xFFFFFFFFull);                   // tensor_dim0_stride[31:0]
  g1[6] = (int)((stride0 >> 32) & 0xFFFFull);               // stride[47:32], stride1=0
  g1[7] = 0;
  i32x4 z4 = {0,0,0,0};
#if defined(__clang_major__) && (__clang_major__ >= 23)
  i32x8 z8 = {0,0,0,0,0,0,0,0};
  __builtin_amdgcn_tensor_load_to_lds(g0, g1, z4, z4, z8, 0);
#else
  __builtin_amdgcn_tensor_load_to_lds(g0, g1, z4, z4, 0);
#endif
}
#endif // HAVE_TDM

// ---------------- fp32 -> bf16 convert ----------------
__global__ __launch_bounds__(256) void cvt_f32_bf16(const float* __restrict__ in,
                                                    bf16_t* __restrict__ out, int n4) {
  int i = blockIdx.x * 256 + threadIdx.x;
  if (i < n4) {
    float4 f = ((const float4*)in)[i];
    out[i*4+0] = (bf16_t)f.x;
    out[i*4+1] = (bf16_t)f.y;
    out[i*4+2] = (bf16_t)f.z;
    out[i*4+3] = (bf16_t)f.w;
  }
}

// ---------------- Generic tiled WMMA GEMM:  C[M,N] = A[M,K] * W[N,K]^T ----------------
// block tile 128x128, 256 threads = 8 waves in 4(m) x 2(n); wave tile 32x64 = 2x4 WMMA tiles.
// K-tiles double-buffered and fed by the Tensor Data Mover (TENSORcnt pipelining).
enum { EPI_BF16 = 0, EPI_BIAS_GELU_BF16 = 1, EPI_BIAS_F32 = 2 };

template<int EPI>
__global__ __launch_bounds__(256) void gemm_kernel(const bf16_t* __restrict__ A,
                                                   const bf16_t* __restrict__ W,
                                                   const float*  __restrict__ bias,
                                                   float*  __restrict__ Cf,
                                                   bf16_t* __restrict__ Cb,
                                                   int K, int ldc) {
  __shared__ bf16_t As[2][128][40];   // 32 data + 8 pad; row stride 80B (16B multiple)
  __shared__ bf16_t Ws[2][128][40];

  const int tid  = threadIdx.x;
  const int lane = tid & 31;
  const int w    = tid >> 5;
  const int wm   = w >> 1;     // 0..3
  const int wn   = w & 1;      // 0..1

  const bf16_t* Arow = A + (size_t)(blockIdx.y * 128) * K;
  const bf16_t* Wrow = W + (size_t)(blockIdx.x * 128) * K;

  v8f acc[2][4];
  #pragma unroll
  for (int i = 0; i < 2; i++)
    #pragma unroll
    for (int j = 0; j < 4; j++) acc[i][j] = v8f_zero();

  const int nsteps = K >> 5;

#if HAVE_TDM
  // LDS rows: 32 bf16 = 16 DWORDs data (pad_interval code 3), pad 4 DWORDs (code 3).
  if (w == 0) {
    tdm_load_2d(Arow, lds_off(&As[0][0][0]), 32, 128, (unsigned long long)K, K, 128, 3, 3);
    tdm_load_2d(Wrow, lds_off(&Ws[0][0][0]), 32, 128, (unsigned long long)K, K, 128, 3, 3);
  }
  for (int s = 0; s < nsteps; s++) {
    const int cur = s & 1;
    if (w == 0) {
      if (s + 1 < nsteps) {
        tdm_load_2d(Arow + (s + 1) * 32, lds_off(&As[cur ^ 1][0][0]), 32, 128,
                    (unsigned long long)K, K, 128, 3, 3);
        tdm_load_2d(Wrow + (s + 1) * 32, lds_off(&Ws[cur ^ 1][0][0]), 32, 128,
                    (unsigned long long)K, K, 128, 3, 3);
        __builtin_amdgcn_s_wait_tensorcnt(2);   // retire current buffer's 2 loads
      } else {
        __builtin_amdgcn_s_wait_tensorcnt(0);
      }
    }
    __syncthreads();                            // tiles[cur] ready for everyone

    v16bf af[2], bfg[4];
    #pragma unroll
    for (int mt = 0; mt < 2; mt++)
      af[mt] = frag_a(&As[cur][wm*32 + mt*16 + (lane & 15)][0], lane);
    #pragma unroll
    for (int nt = 0; nt < 4; nt++)
      bfg[nt] = frag_b(&Ws[cur][wn*64 + nt*16 + (lane & 15)][0], lane);

    #pragma unroll
    for (int mt = 0; mt < 2; mt++)
      #pragma unroll
      for (int nt = 0; nt < 4; nt++)
        acc[mt][nt] = wmma_bf16(af[mt], bfg[nt], acc[mt][nt]);

    __syncthreads();                            // reads done before TDM reuses buffer
  }
#else
  for (int s = 0; s < nsteps; s++) {
    const int kb = s * 32;
    __syncthreads();
    #pragma unroll
    for (int i = 0; i < 2; i++) {
      int c  = tid * 2 + i;
      int r  = c >> 2;
      int kc = (c & 3) << 3;
      *(uint4*)&As[0][r][kc] = *(const uint4*)(Arow + (size_t)r * K + kb + kc);
      *(uint4*)&Ws[0][r][kc] = *(const uint4*)(Wrow + (size_t)r * K + kb + kc);
      if (kb + 32 < K) {
        __builtin_prefetch(Arow + (size_t)r * K + kb + 32 + kc, 0, 1);
        __builtin_prefetch(Wrow + (size_t)r * K + kb + 32 + kc, 0, 1);
      }
    }
    __syncthreads();
    v16bf af[2], bfg[4];
    #pragma unroll
    for (int mt = 0; mt < 2; mt++)
      af[mt] = frag_a(&As[0][wm*32 + mt*16 + (lane & 15)][0], lane);
    #pragma unroll
    for (int nt = 0; nt < 4; nt++)
      bfg[nt] = frag_b(&Ws[0][wn*64 + nt*16 + (lane & 15)][0], lane);
    #pragma unroll
    for (int mt = 0; mt < 2; mt++)
      #pragma unroll
      for (int nt = 0; nt < 4; nt++)
        acc[mt][nt] = wmma_bf16(af[mt], bfg[nt], acc[mt][nt]);
  }
#endif

  const int half = lane >> 4, nl = lane & 15;
  #pragma unroll
  for (int mt = 0; mt < 2; mt++)
    #pragma unroll
    for (int nt = 0; nt < 4; nt++)
      #pragma unroll
      for (int r = 0; r < 8; r++) {
        int m = blockIdx.y*128 + wm*32 + mt*16 + r + 8*half;
        int n = blockIdx.x*128 + wn*64 + nt*16 + nl;
        float v = acc[mt][nt][r];
        if constexpr (EPI == EPI_BIAS_GELU_BF16) v = gelu_exact(v + bias[n]);
        if constexpr (EPI == EPI_BIAS_F32)       v = v + bias[n];
        if constexpr (EPI == EPI_BIAS_F32) Cf[(size_t)m * ldc + n] = v;
        else                               Cb[(size_t)m * ldc + n] = (bf16_t)v;
      }
}

// ---------------- RoPE in place on packed qkv [M, 6144] (q,k sections) ----------------
__global__ __launch_bounds__(256) void rope_kernel(bf16_t* __restrict__ qkv,
                                                   const float* __restrict__ cosp,
                                                   const float* __restrict__ sinp) {
  int idx = blockIdx.x * 256 + threadIdx.x;  // over B*S*H*64
  int d = idx & 63;
  int h = (idx >> 6) & (N_HEADS - 1);
  int s = (idx >> 10) & (SS - 1);
  int b = idx >> 20;
  const float c1 = cosp[s*HEAD_DIM + d],      s1 = sinp[s*HEAD_DIM + d];
  const float c2 = cosp[s*HEAD_DIM + d + 64], s2 = sinp[s*HEAD_DIM + d + 64];
  size_t base = ((size_t)(b*SS + s)) * QKV_N + h * HEAD_DIM;
  #pragma unroll
  for (int proj = 0; proj < 2; proj++) {
    bf16_t* p = qkv + base + proj * D_MODEL;
    float t1 = (float)p[d], t2 = (float)p[d + 64];
    p[d]      = (bf16_t)(t1 * c1 - t2 * s1);   // rotate_half: first half gets -t2*sin
    p[d + 64] = (bf16_t)(t2 * c2 + t1 * s2);
  }
}

// ---------------- Flash attention: one (b,h,q-tile) per block ----------------
__global__ __launch_bounds__(256) void attn_kernel(const bf16_t* __restrict__ qkv,
                                                   float* __restrict__ out) {
  const int qt = blockIdx.x, h = blockIdx.y, b = blockIdx.z;
  const size_t rs = QKV_N;
  const bf16_t* qbase = qkv + ((size_t)b * SS) * rs + h * HEAD_DIM;
  const bf16_t* kbase = qbase + D_MODEL;
  const bf16_t* vbase = qbase + 2 * D_MODEL;

  __shared__ bf16_t Qs[128][136];
  __shared__ bf16_t Ks[128][136];   // reused as P (bf16 probabilities) after scores consumed
  __shared__ bf16_t Vt[128][136];   // V transposed: [dh][key]

  const int tid = threadIdx.x, lane = tid & 31, w = tid >> 5;
  const int half = lane >> 4, nl = lane & 15;

  // Load 128x128 Q tile. LDS rows: 128 bf16 = 64 DWORDs (pad_interval code 5), pad 4 DWORDs (code 3).
#if HAVE_TDM
  if (w == 0) {
    tdm_load_2d(qbase + (size_t)(qt*128) * rs, lds_off(&Qs[0][0]), 128, 128,
                (unsigned long long)rs, 128, 128, 5, 3);
    __builtin_amdgcn_s_wait_tensorcnt(0);
  }
#else
  #pragma unroll
  for (int i = 0; i < 8; i++) {
    int c = tid + i * 256;           // 0..2047
    int r = c >> 4, col = (c & 15) << 3;
    *(uint4*)&Qs[r][col] = *(const uint4*)(qbase + (size_t)(qt*128 + r) * rs + col);
  }
#endif
  __syncthreads();

  v16bf qf[4];
  #pragma unroll
  for (int kc = 0; kc < 4; kc++)
    qf[kc] = frag_a(&Qs[w*16 + nl][kc*32], lane);

  v8f Oacc[8];
  float mrow[8], lrow[8];
  #pragma unroll
  for (int i = 0; i < 8; i++) { Oacc[i] = v8f_zero(); mrow[i] = -1e30f; lrow[i] = 0.f; }
  const float scale = 0.08838834764831845f;   // 1/sqrt(128)

  for (int kt = 0; kt < 8; kt++) {
    __syncthreads();   // everyone finished with previous Ks(P)/Vt tiles
#if HAVE_TDM
    if (w == 0)
      tdm_load_2d(kbase + (size_t)(kt*128) * rs, lds_off(&Ks[0][0]), 128, 128,
                  (unsigned long long)rs, 128, 128, 5, 3);
    // V transpose (manual) overlaps with the in-flight K DMA
    #pragma unroll
    for (int i = 0; i < 8; i++) {
      int c = tid + i * 256;
      int r = c >> 4, col = (c & 15) << 3;
      union { uint4 u; bf16_t e[8]; } t;
      t.u = *(const uint4*)(vbase + (size_t)(kt*128 + r) * rs + col);
      #pragma unroll
      for (int j = 0; j < 8; j++) Vt[col + j][r] = t.e[j];
    }
    if (w == 0) __builtin_amdgcn_s_wait_tensorcnt(0);
#else
    #pragma unroll
    for (int i = 0; i < 8; i++) {
      int c = tid + i * 256;
      int r = c >> 4, col = (c & 15) << 3;
      *(uint4*)&Ks[r][col] = *(const uint4*)(kbase + (size_t)(kt*128 + r) * rs + col);
      union { uint4 u; bf16_t e[8]; } t;
      t.u = *(const uint4*)(vbase + (size_t)(kt*128 + r) * rs + col);
      #pragma unroll
      for (int j = 0; j < 8; j++) Vt[col + j][r] = t.e[j];
    }
#endif
    __syncthreads();

    // Scores: 16 q-rows x 128 keys per wave
    v8f sc[8];
    #pragma unroll
    for (int nt = 0; nt < 8; nt++) {
      v8f a = v8f_zero();
      #pragma unroll
      for (int kc = 0; kc < 4; kc++) {
        v16bf bfr = frag_b(&Ks[nt*16 + nl][kc*32], lane);
        a = wmma_bf16(qf[kc], bfr, a);
      }
      sc[nt] = a;
    }

    // Online softmax update (mask is all-true)
    float fac[8];
    #pragma unroll
    for (int r = 0; r < 8; r++) {
      float v = -1e30f;
      #pragma unroll
      for (int nt = 0; nt < 8; nt++) { sc[nt][r] *= scale; v = fmaxf(v, sc[nt][r]); }
      #pragma unroll
      for (int off = 1; off < 16; off <<= 1) v = fmaxf(v, __shfl_xor(v, off));
      float mnew = fmaxf(mrow[r], v);
      fac[r] = __expf(mrow[r] - mnew);
      float sum = 0.f;
      #pragma unroll
      for (int nt = 0; nt < 8; nt++) { float p = __expf(sc[nt][r] - mnew); sc[nt][r] = p; sum += p; }
      #pragma unroll
      for (int off = 1; off < 16; off <<= 1) sum += __shfl_xor(sum, off);
      lrow[r] = lrow[r] * fac[r] + sum;
      mrow[r] = mnew;
    }
    #pragma unroll
    for (int ot = 0; ot < 8; ot++)
      #pragma unroll
      for (int r = 0; r < 8; r++) Oacc[ot][r] *= fac[r];

    // Stash P (bf16) into Ks overlay — wave-private 16-row region
    __syncthreads();
    #pragma unroll
    for (int nt = 0; nt < 8; nt++)
      #pragma unroll
      for (int r = 0; r < 8; r++)
        Ks[w*16 + r + 8*half][nt*16 + nl] = (bf16_t)sc[nt][r];

    // O += P @ V
    #pragma unroll
    for (int kc = 0; kc < 4; kc++) {
      v16bf pf = frag_a(&Ks[w*16 + nl][kc*32], lane);
      #pragma unroll
      for (int ot = 0; ot < 8; ot++) {
        v16bf vf = frag_b(&Vt[ot*16 + nl][kc*32], lane);
        Oacc[ot] = wmma_bf16(pf, vf, Oacc[ot]);
      }
    }
  }

  // Finalize: divide by row sums, write f32 [B,S,D] head-major
  #pragma unroll
  for (int ot = 0; ot < 8; ot++)
    #pragma unroll
    for (int r = 0; r < 8; r++) {
      int m  = qt*128 + w*16 + r + 8*half;
      int dh = ot*16 + nl;
      out[((size_t)(b*SS) + m) * D_MODEL + h*HEAD_DIM + dh] = Oacc[ot][r] / lrow[r];
    }
}

// ---------------- residual add + LayerNorm (row of 2048) ----------------
__global__ __launch_bounds__(256) void add_ln_kernel(const float* __restrict__ a,
                                                     const float* __restrict__ b,
                                                     const float* __restrict__ scale,
                                                     const float* __restrict__ bias,
                                                     float* __restrict__ of,
                                                     bf16_t* __restrict__ ob) {
  __shared__ float red[256];
  const int row = blockIdx.x, tid = threadIdx.x;
  const size_t base = (size_t)row * D_MODEL;
  float v[8]; float s1 = 0.f;
  #pragma unroll
  for (int i = 0; i < 8; i++) { int c = tid + i*256; v[i] = a[base+c] + b[base+c]; s1 += v[i]; }
  red[tid] = s1; __syncthreads();
  for (int off = 128; off > 0; off >>= 1) { if (tid < off) red[tid] += red[tid + off]; __syncthreads(); }
  float mean = red[0] * (1.0f / D_MODEL); __syncthreads();
  float s2 = 0.f;
  #pragma unroll
  for (int i = 0; i < 8; i++) { float d = v[i] - mean; s2 += d * d; }
  red[tid] = s2; __syncthreads();
  for (int off = 128; off > 0; off >>= 1) { if (tid < off) red[tid] += red[tid + off]; __syncthreads(); }
  float inv = rsqrtf(red[0] * (1.0f / D_MODEL) + 1e-5f);
  #pragma unroll
  for (int i = 0; i < 8; i++) {
    int c = tid + i*256;
    float y = (v[i] - mean) * inv * scale[c] + bias[c];
    of[base + c] = y;
    if (ob) ob[base + c] = (bf16_t)y;
  }
}

// ---------------- host orchestration ----------------
extern "C" void kernel_launch(void* const* d_in, const int* in_sizes, int n_in,
                              void* d_out, int out_size, void* d_ws, size_t ws_size,
                              hipStream_t stream) {
  const float* x    = (const float*)d_in[0];
  const float* cosp = (const float*)d_in[1];
  const float* sinp = (const float*)d_in[2];
  // d_in[3] = mask (all true) — unused
  const float* Wqkv = (const float*)d_in[4];
  const float* ln1s = (const float*)d_in[5];
  const float* ln1b = (const float*)d_in[6];
  const float* W1   = (const float*)d_in[7];
  const float* b1   = (const float*)d_in[8];
  const float* W2   = (const float*)d_in[9];
  const float* b2   = (const float*)d_in[10];
  const float* ln2s = (const float*)d_in[11];
  const float* ln2b = (const float*)d_in[12];
  float* out = (float*)d_out;

  char* ws = (char*)d_ws;
  size_t off = 0;
  auto alloc = [&](size_t bytes) { void* p = ws + off; off += (bytes + 255) & ~(size_t)255; return p; };

  bf16_t* xb    = (bf16_t*)alloc((size_t)MROWS * D_MODEL * 2);
  bf16_t* wqkvb = (bf16_t*)alloc((size_t)QKV_N * D_MODEL * 2);
  bf16_t* w1b   = (bf16_t*)alloc((size_t)FF_DIM * D_MODEL * 2);
  bf16_t* w2b   = (bf16_t*)alloc((size_t)D_MODEL * FF_DIM * 2);
  bf16_t* qkvb  = (bf16_t*)alloc((size_t)MROWS * QKV_N * 2);
  float*  attn  = (float*)alloc((size_t)MROWS * D_MODEL * 4);
  float*  x1    = (float*)alloc((size_t)MROWS * D_MODEL * 4);
  bf16_t* x1b   = (bf16_t*)alloc((size_t)MROWS * D_MODEL * 2);
  bf16_t* hb    = (bf16_t*)alloc((size_t)MROWS * FF_DIM * 2);
  float*  ff    = (float*)alloc((size_t)MROWS * D_MODEL * 4);

  // 1) fp32 -> bf16 conversions
  cvt_f32_bf16<<<(MROWS*D_MODEL/4 + 255)/256, 256, 0, stream>>>(x, xb, MROWS*D_MODEL/4);
  cvt_f32_bf16<<<(QKV_N*D_MODEL/4 + 255)/256, 256, 0, stream>>>(Wqkv, wqkvb, QKV_N*D_MODEL/4);
  cvt_f32_bf16<<<(FF_DIM*D_MODEL/4 + 255)/256, 256, 0, stream>>>(W1, w1b, FF_DIM*D_MODEL/4);
  cvt_f32_bf16<<<(D_MODEL*FF_DIM/4 + 255)/256, 256, 0, stream>>>(W2, w2b, D_MODEL*FF_DIM/4);

  // 2) QKV projection: [8192,2048] x [6144,2048]^T -> bf16 [8192,6144]
  gemm_kernel<EPI_BF16><<<dim3(QKV_N/128, MROWS/128), 256, 0, stream>>>(
      xb, wqkvb, nullptr, nullptr, qkvb, D_MODEL, QKV_N);

  // 3) RoPE in place on q,k sections
  rope_kernel<<<(BB*SS*N_HEADS*64)/256, 256, 0, stream>>>(qkvb, cosp, sinp);

  // 4) Flash attention
  attn_kernel<<<dim3(SS/128, N_HEADS, BB), 256, 0, stream>>>(qkvb, attn);

  // 5) x1 = LN(x + attn)
  add_ln_kernel<<<MROWS, 256, 0, stream>>>(x, attn, ln1s, ln1b, x1, x1b);

  // 6) h = gelu(x1 @ W1^T + b1) -> bf16 [8192,8192]
  gemm_kernel<EPI_BIAS_GELU_BF16><<<dim3(FF_DIM/128, MROWS/128), 256, 0, stream>>>(
      x1b, w1b, b1, nullptr, hb, D_MODEL, FF_DIM);

  // 7) ff = h @ W2^T + b2 -> f32 [8192,2048]
  gemm_kernel<EPI_BIAS_F32><<<dim3(D_MODEL/128, MROWS/128), 256, 0, stream>>>(
      hb, w2b, b2, ff, nullptr, FF_DIM, D_MODEL);

  // 8) out = LN(x1 + ff)
  add_ln_kernel<<<MROWS, 256, 0, stream>>>(x1, ff, ln2s, ln2b, out, nullptr);
}
